// ScaffoldGAN_73400991089362
// MI455X (gfx1250) — compile-verified, matrix operand order checked by reading, not run on hardware
//
#include <hip/hip_runtime.h>
#include <cstdint>

// ---------------- problem constants (from reference) ----------------
#define Hdim 450
#define NNODES 4000
#define MMESS 6000
#define KNEI 6
#define BROOT 256
#define DHdim 450
#define DEPTH_C 15

// ---------------- padded dims for WMMA tiling ----------------
#define HP 480          // feature dim padded: divisible by 32 (K step) and 16
#define K2 960          // 2*HP for concat GEMM (discriminator layer 1)
#define WR 512          // weight rows padded: 4 col-blocks of 128
#define MP 6016         // MMESS padded to 128 (47 tiles)
#define MKP 36096       // MMESS*KNEI padded to 128 (282 tiles)

#if defined(__has_builtin)
#if __has_builtin(__builtin_amdgcn_global_load_async_to_lds_b128)
#define USE_ASYNC_LDS 1
#endif
#endif
#ifndef USE_ASYNC_LDS
#define USE_ASYNC_LDS 0
#endif

typedef __attribute__((ext_vector_type(16))) __bf16 v16bf;
typedef __attribute__((ext_vector_type(8)))  float  v8f;

#if USE_ASYNC_LDS
// exact parameter types of __builtin_amdgcn_global_load_async_to_lds_b128:
// (int4 addrspace(1)*, int4 addrspace(3)*, imm offset, imm cpol)
typedef int v4i_vs __attribute__((__vector_size__(16)));
typedef __attribute__((address_space(1))) v4i_vs* g_v4i_ptr;
typedef __attribute__((address_space(3))) v4i_vs* l_v4i_ptr;
#endif

union FragBF {
  v16bf v;
  uint4 u[2];
};

__device__ __forceinline__ unsigned short f2bf(float x) {
  unsigned int u = __float_as_uint(x);
  u += 0x7fffu + ((u >> 16) & 1u);   // round-to-nearest-even
  return (unsigned short)(u >> 16);
}
__device__ __forceinline__ float bf2f(unsigned short s) {
  return __uint_as_float(((unsigned int)s) << 16);
}

// 16-byte global->LDS copy: async on CDNA5 (ASYNCcnt), sync fallback otherwise.
// AS1 pointer built by inttoptr of the 64-bit flat address; AS3 pointer is the
// low 32 bits of the flat LDS address (== LDS byte offset per ISA aperture rule).
__device__ __forceinline__ void copy16_g2l(const unsigned short* g, unsigned short* l) {
#if USE_ASYNC_LDS
  __builtin_amdgcn_global_load_async_to_lds_b128(
      (g_v4i_ptr)(unsigned long long)(uintptr_t)g,
      (l_v4i_ptr)(unsigned int)(uintptr_t)l,
      0, 0);
#else
  *reinterpret_cast<uint4*>(l) = *reinterpret_cast<const uint4*>(g);
#endif
}

// Stage a 128x32-half A tile and 128x32-half B tile into LDS (8KB each).
// 256 threads: 2 x 16B per thread per tile. LDS rows are 64B (bank-friendly).
__device__ __forceinline__ void stage_tile(const unsigned short* __restrict__ gA, int ldA,
                                           const unsigned short* __restrict__ gB, int ldB,
                                           int k0, unsigned short* lA, unsigned short* lB,
                                           int tid) {
#pragma unroll
  for (int l = 0; l < 2; ++l) {
    const int idx = tid + l * 256;
    const int row = idx >> 2;
    const int c = (idx & 3) * 8;                 // offset in halfs, 8 halfs = 16B
    copy16_g2l(gA + (size_t)row * ldA + k0 + c, lA + row * 32 + c);
    copy16_g2l(gB + (size_t)row * ldB + k0 + c, lB + row * 32 + c);
  }
}

// Read a 16x32 bf16 fragment from an LDS tile (tile rows = 32 halfs).
// Lane r=lane%16 reads row r, two 16B chunks at k=kc and kc+16, kc=(lane/16)*8.
__device__ __forceinline__ v16bf lds_frag(const unsigned short* base) {
  const int lane = threadIdx.x & 31;
  const unsigned short* p = base + (lane & 15) * 32 + ((lane >> 4) << 3);
  FragBF f;
  f.u[0] = *reinterpret_cast<const uint4*>(p);
  f.u[1] = *reinterpret_cast<const uint4*>(p + 16);
  return f.v;
}

// ---------------- bf16 WMMA GEMM: C = act(A @ B^T + addend + bias) ----------------
// A: [Mpad, ldA] bf16 row-major; B: [Npad, ldB] bf16 row-major (torch [out,in]).
// 256 threads = 8 waves (2x4); block tile 128x128; each wave 64x32 via 4x2 fragments.
// K-slab double-buffered through LDS with async global->LDS copies.
__global__ void __launch_bounds__(256)
gemm_bf16_wmma(const unsigned short* __restrict__ A,
               const unsigned short* __restrict__ B,
               float* __restrict__ C,                 // f32 out (or null)
               unsigned short* __restrict__ Cb,       // bf16 out (or null)
               int Mvalid, int Nvalid, int padN,
               int ldA, int ldB, int ldC, int kTiles,
               const float* __restrict__ addend, int ldAdd, int rowDiv,
               const float* __restrict__ bias, int act)
{
  __shared__ unsigned short sA[2][128 * 32];
  __shared__ unsigned short sB[2][128 * 32];

  const int tid = threadIdx.x;
  const int wave = tid >> 5;
  const int lane = tid & 31;
  const int wm = wave & 1;        // 2 row groups of 64
  const int wn = wave >> 1;       // 4 col groups of 32
  const int rowBlk = blockIdx.x * 128;
  const int colBlk = blockIdx.y * 128;

  const unsigned short* gA = A + (size_t)rowBlk * ldA;
  const unsigned short* gB = B + (size_t)colBlk * ldB;

  v8f acc[4][2] = {};

  stage_tile(gA, ldA, gB, ldB, 0, sA[0], sB[0], tid);

  for (int kt = 0; kt < kTiles; ++kt) {
    const int cur = kt & 1;
    if (kt + 1 < kTiles) {
      stage_tile(gA, ldA, gB, ldB, (kt + 1) * 32, sA[cur ^ 1], sB[cur ^ 1], tid);
#if USE_ASYNC_LDS
      asm volatile("s_wait_asynccnt 0x4" ::: "memory");   // slab kt landed; kt+1 in flight
#endif
    } else {
#if USE_ASYNC_LDS
      asm volatile("s_wait_asynccnt 0x0" ::: "memory");
#endif
    }
    __syncthreads();

    v16bf af[4], bfr[2];
#pragma unroll
    for (int i = 0; i < 4; ++i) af[i] = lds_frag(sA[cur] + (wm * 64 + i * 16) * 32);
#pragma unroll
    for (int j = 0; j < 2; ++j) bfr[j] = lds_frag(sB[cur] + (wn * 32 + j * 16) * 32);
#pragma unroll
    for (int i = 0; i < 4; ++i)
#pragma unroll
      for (int j = 0; j < 2; ++j)
        acc[i][j] = __builtin_amdgcn_wmma_f32_16x16x32_bf16(
            false, af[i], false, bfr[j], (short)0, acc[i][j], false, false);

    __syncthreads();   // tile fully consumed before its buffer is overwritten
  }

  // C/D layout: lane<16 -> N=lane, M=v ; lane>=16 -> N=lane-16, M=8+v
  const int cn = lane & 15;
  const int mhalf = (lane >> 4) << 3;
#pragma unroll
  for (int i = 0; i < 4; ++i) {
#pragma unroll
    for (int j = 0; j < 2; ++j) {
      const int col = colBlk + wn * 32 + j * 16 + cn;
      const float bv = (bias && col < Nvalid) ? bias[col] : 0.0f;
#pragma unroll
      for (int v = 0; v < 8; ++v) {
        const int row = rowBlk + wm * 64 + i * 16 + mhalf + v;
        if (row >= Mvalid) continue;
        if (col < Nvalid) {
          float val = acc[i][j][v] + bv;
          if (addend) val += addend[(size_t)(row / rowDiv) * ldAdd + col];
          if (act == 1)      val = 1.0f / (1.0f + __expf(-val));    // sigmoid
          else if (act == 2) val = tanhf(val);                      // tanh
          else if (act == 3) val = (val > 0.0f) ? val : 0.1f * val; // leaky 0.1
          if (C)  C [(size_t)row * ldC + col] = val;
          if (Cb) Cb[(size_t)row * ldC + col] = f2bf(val);
        } else if (Cb && col < padN) {
          Cb[(size_t)row * ldC + col] = 0;  // keep bf16 pad columns zero
        }
      }
    }
  }
}

// ---------------- small helper kernels ----------------

__global__ void zero_f32(float* p, int n) {
  int t = blockIdx.x * blockDim.x + threadIdx.x;
  if (t < n) p[t] = 0.0f;
}

// fnode_e[n][j] = emb[fnode[n]][j]
__global__ void embed_nodes(const int* __restrict__ fnode, const float* __restrict__ emb,
                            float* __restrict__ fnode_e) {
  int t = blockIdx.x * blockDim.x + threadIdx.x;
  if (t >= NNODES * Hdim) return;
  int n = t / Hdim, j = t - n * Hdim;
  fnode_e[t] = emb[(size_t)fnode[n] * Hdim + j];
}

// x_bf[m][j] = bf16(emb[fnode[fmess[m]]][j]), zero-padded to HP
__global__ void build_x(const int* __restrict__ fmess, const int* __restrict__ fnode,
                        const float* __restrict__ emb, unsigned short* __restrict__ x_bf) {
  __shared__ int node;
  int m = blockIdx.x;
  if (threadIdx.x == 0) node = fnode[fmess[m]];
  __syncthreads();
  const float* src = emb + (size_t)node * Hdim;
  for (int j = threadIdx.x; j < HP; j += blockDim.x)
    x_bf[(size_t)m * HP + j] = f2bf(j < Hdim ? src[j] : 0.0f);
}

// Pack f32 matrix slice into zero-padded bf16 [Nrows, dstLd]
__global__ void pack_bf(unsigned short* __restrict__ dst, int dstLd, int kCount,
                        const float* __restrict__ src, int srcLd, int koff,
                        int Nv, int Kv, int Nrows) {
  int t = blockIdx.x * blockDim.x + threadIdx.x;
  if (t >= Nrows * kCount) return;
  int n = t / kCount, k = t - n * kCount;
  float v = (n < Nv && k < Kv) ? src[(size_t)n * srcLd + koff + k] : 0.0f;
  dst[(size_t)n * dstLd + k] = f2bf(v);
}

// Per step: h_nei gather + sum over K; emit sum_h (f32 + bf16) and h_nei (bf16)
__global__ void gather_step(const float* __restrict__ h, const int* __restrict__ mess_graph,
                            float* __restrict__ sumh_f, unsigned short* __restrict__ sumh_bf,
                            unsigned short* __restrict__ hnei_bf) {
  __shared__ int idx[KNEI];
  int m = blockIdx.x;
  if (threadIdx.x < KNEI) idx[threadIdx.x] = mess_graph[m * KNEI + threadIdx.x];
  __syncthreads();
  for (int j = threadIdx.x; j < HP; j += blockDim.x) {
    float s = 0.0f;
#pragma unroll
    for (int k = 0; k < KNEI; ++k) {
      float v = (j < Hdim) ? h[(size_t)idx[k] * Hdim + j] : 0.0f;
      hnei_bf[((size_t)(m * KNEI + k)) * HP + j] = f2bf(v);
      s += v;
    }
    if (j < Hdim) sumh_f[(size_t)m * Hdim + j] = s;
    sumh_bf[(size_t)m * HP + j] = f2bf(s);
  }
}

// sum_gh[m][j] = sum_k r[m*K+k][j] * h[mess_graph[m][k]][j]  (bf16 out, padded)
__global__ void sum_gh_kernel(const unsigned short* __restrict__ r_bf, const float* __restrict__ h,
                              const int* __restrict__ mess_graph,
                              unsigned short* __restrict__ sumgh_bf) {
  __shared__ int idx[KNEI];
  int m = blockIdx.x;
  if (threadIdx.x < KNEI) idx[threadIdx.x] = mess_graph[m * KNEI + threadIdx.x];
  __syncthreads();
  for (int j = threadIdx.x; j < HP; j += blockDim.x) {
    float s = 0.0f;
    if (j < Hdim) {
#pragma unroll
      for (int k = 0; k < KNEI; ++k)
        s += bf2f(r_bf[((size_t)(m * KNEI + k)) * HP + j]) * h[(size_t)idx[k] * Hdim + j];
    }
    sumgh_bf[(size_t)m * HP + j] = f2bf(s);
  }
}

// h = ((1-z)*sum_h + z*pre_h) * mask   (mask kills message 0)
__global__ void combine(float* __restrict__ h, const float* __restrict__ z,
                        const float* __restrict__ sumh, const float* __restrict__ preh) {
  int t = blockIdx.x * blockDim.x + threadIdx.x;
  if (t >= MMESS * Hdim) return;
  float zz = z[t];
  float val = (1.0f - zz) * sumh[t] + zz * preh[t];
  if (t < Hdim) val = 0.0f;   // row m==0 masked
  h[t] = val;
}

// mess_nei[n][j] = sum_k h[node_graph[n][k]][j]
__global__ void node_agg(const float* __restrict__ h, const int* __restrict__ node_graph,
                         float* __restrict__ mess_nei) {
  __shared__ int idx[KNEI];
  int n = blockIdx.x;
  if (threadIdx.x < KNEI) idx[threadIdx.x] = node_graph[n * KNEI + threadIdx.x];
  __syncthreads();
  for (int j = threadIdx.x; j < Hdim; j += blockDim.x) {
    float s = 0.0f;
#pragma unroll
    for (int k = 0; k < KNEI; ++k) s += h[(size_t)idx[k] * Hdim + j];
    mess_nei[(size_t)n * Hdim + j] = s;
  }
}

// root_bf[b] = bf16 of [fnode_e[root], mess_nei[root]] padded into [HP | HP]
__global__ void root_pack(const int* __restrict__ root_idx, const float* __restrict__ fnode_e,
                          const float* __restrict__ mess_nei, unsigned short* __restrict__ root_bf) {
  int b = blockIdx.x;
  int rn = root_idx[b];
  for (int j = threadIdx.x; j < K2; j += blockDim.x) {
    float v = 0.0f;
    if (j < HP) { if (j < Hdim) v = fnode_e[(size_t)rn * Hdim + j]; }
    else { int jj = j - HP; if (jj < Hdim) v = mess_nei[(size_t)rn * Hdim + jj]; }
    root_bf[(size_t)b * K2 + j] = f2bf(v);
  }
}

// score[b] = h2[b] . D3_w + D3_b   (one wave per root)
__global__ void final_dot(const float* __restrict__ h2, const float* __restrict__ D3w,
                          const float* __restrict__ D3b, float* __restrict__ out) {
  int b = blockIdx.x;
  float s = 0.0f;
  for (int j = threadIdx.x; j < DHdim; j += 32) s += h2[(size_t)b * DHdim + j] * D3w[j];
#pragma unroll
  for (int off = 16; off > 0; off >>= 1) s += __shfl_down(s, off, 32);
  if (threadIdx.x == 0) out[b] = s + D3b[0];
}

// ---------------- host orchestration ----------------
extern "C" void kernel_launch(void* const* d_in, const int* in_sizes, int n_in,
                              void* d_out, int out_size, void* d_ws, size_t ws_size,
                              hipStream_t stream) {
  const int*   fnode      = (const int*)  d_in[0];
  const int*   fmess      = (const int*)  d_in[1];
  const int*   node_graph = (const int*)  d_in[2];
  const int*   mess_graph = (const int*)  d_in[3];
  const int*   root_idx   = (const int*)  d_in[4];
  const float* emb        = (const float*)d_in[5];
  const float* W_z_w      = (const float*)d_in[6];
  const float* W_z_b      = (const float*)d_in[7];
  const float* W_r_w      = (const float*)d_in[8];
  const float* U_r_w      = (const float*)d_in[9];
  const float* U_r_b      = (const float*)d_in[10];
  const float* W_h_w      = (const float*)d_in[11];
  const float* W_h_b      = (const float*)d_in[12];
  const float* D1_w       = (const float*)d_in[13];
  const float* D1_b       = (const float*)d_in[14];
  const float* D2_w       = (const float*)d_in[15];
  const float* D2_b       = (const float*)d_in[16];
  const float* D3_w       = (const float*)d_in[17];
  const float* D3_b       = (const float*)d_in[18];
  // depth (d_in[19]) is a fixed architectural constant: DEPTH_C == 15

  char* ws = (char*)d_ws;
  size_t off = 0;
  auto alloc = [&](size_t bytes) -> void* {
    off = (off + 255) & ~(size_t)255;
    void* p = ws + off;
    off += bytes;
    return p;
  };

  float*          fnode_e  = (float*)         alloc((size_t)NNODES * Hdim * 4);
  unsigned short* x_bf     = (unsigned short*)alloc((size_t)MP * HP * 2);
  unsigned short* Wz_x     = (unsigned short*)alloc((size_t)WR * HP * 2);
  unsigned short* Wz_h     = (unsigned short*)alloc((size_t)WR * HP * 2);
  unsigned short* Wh_x     = (unsigned short*)alloc((size_t)WR * HP * 2);
  unsigned short* Wh_h     = (unsigned short*)alloc((size_t)WR * HP * 2);
  unsigned short* Wr       = (unsigned short*)alloc((size_t)WR * HP * 2);
  unsigned short* Ur       = (unsigned short*)alloc((size_t)WR * HP * 2);
  unsigned short* D1_bf    = (unsigned short*)alloc((size_t)WR * K2 * 2);
  unsigned short* D2_bf    = (unsigned short*)alloc((size_t)WR * HP * 2);
  float*          zx       = (float*)         alloc((size_t)MMESS * Hdim * 4);
  float*          hx       = (float*)         alloc((size_t)MMESS * Hdim * 4);
  float*          rx       = (float*)         alloc((size_t)MMESS * Hdim * 4);
  float*          h        = (float*)         alloc((size_t)MMESS * Hdim * 4);
  float*          sumh_f   = (float*)         alloc((size_t)MMESS * Hdim * 4);
  unsigned short* sumh_bf  = (unsigned short*)alloc((size_t)MP * HP * 2);
  unsigned short* hnei_bf  = (unsigned short*)alloc((size_t)MKP * HP * 2);
  unsigned short* r_bf     = (unsigned short*)alloc((size_t)MKP * HP * 2);
  float*          z_buf    = (float*)         alloc((size_t)MMESS * Hdim * 4);
  float*          preh     = (float*)         alloc((size_t)MMESS * Hdim * 4);
  unsigned short* sumgh_bf = (unsigned short*)alloc((size_t)MP * HP * 2);
  float*          mess_nei = (float*)         alloc((size_t)NNODES * Hdim * 4);
  unsigned short* root_bf  = (unsigned short*)alloc((size_t)BROOT * K2 * 2);
  unsigned short* h1_bf    = (unsigned short*)alloc((size_t)BROOT * HP * 2);
  float*          h2_f     = (float*)         alloc((size_t)BROOT * DHdim * 4);

  const dim3 blk128(128);
  const dim3 blk256(256);
  const int KT = HP / 32;      // 15
  const int KT2 = K2 / 32;     // 30
  const dim3 gM(MP / 128, WR / 128);    // 47 x 4
  const dim3 gMK(MKP / 128, WR / 128);  // 282 x 4
  const dim3 gB(BROOT / 128, WR / 128); // 2 x 4

  // ---- one-time setup ----
  zero_f32<<<(MMESS * Hdim + 255) / 256, 256, 0, stream>>>(h, MMESS * Hdim);
  embed_nodes<<<(NNODES * Hdim + 255) / 256, 256, 0, stream>>>(fnode, emb, fnode_e);
  build_x<<<MMESS, blk128, 0, stream>>>(fmess, fnode, emb, x_bf);

  const int PACKN = (WR * HP + 255) / 256;
  pack_bf<<<PACKN, 256, 0, stream>>>(Wz_x, HP, HP, W_z_w, 2 * Hdim, 0,     Hdim, Hdim, WR);
  pack_bf<<<PACKN, 256, 0, stream>>>(Wz_h, HP, HP, W_z_w, 2 * Hdim, Hdim,  Hdim, Hdim, WR);
  pack_bf<<<PACKN, 256, 0, stream>>>(Wh_x, HP, HP, W_h_w, 2 * Hdim, 0,     Hdim, Hdim, WR);
  pack_bf<<<PACKN, 256, 0, stream>>>(Wh_h, HP, HP, W_h_w, 2 * Hdim, Hdim,  Hdim, Hdim, WR);
  pack_bf<<<PACKN, 256, 0, stream>>>(Wr,   HP, HP, W_r_w, Hdim,     0,     Hdim, Hdim, WR);
  pack_bf<<<PACKN, 256, 0, stream>>>(Ur,   HP, HP, U_r_w, Hdim,     0,     Hdim, Hdim, WR);
  pack_bf<<<PACKN, 256, 0, stream>>>(D1_bf,        K2, HP, D1_w, 2 * DHdim, 0,     DHdim, DHdim, WR);
  pack_bf<<<PACKN, 256, 0, stream>>>(D1_bf + HP,   K2, HP, D1_w, 2 * DHdim, DHdim, DHdim, DHdim, WR);
  pack_bf<<<PACKN, 256, 0, stream>>>(D2_bf, HP, HP, D2_w, DHdim, 0, DHdim, DHdim, WR);

  // ---- loop-invariant x-side GEMMs: zx = x@Wz_x^T + bz ; hx = x@Wh_x^T + bh ; rx = x@Wr^T
  gemm_bf16_wmma<<<gM, blk256, 0, stream>>>(x_bf, Wz_x, zx, nullptr, MMESS, Hdim, HP,
                                            HP, HP, Hdim, KT, nullptr, 0, 1, W_z_b, 0);
  gemm_bf16_wmma<<<gM, blk256, 0, stream>>>(x_bf, Wh_x, hx, nullptr, MMESS, Hdim, HP,
                                            HP, HP, Hdim, KT, nullptr, 0, 1, W_h_b, 0);
  gemm_bf16_wmma<<<gM, blk256, 0, stream>>>(x_bf, Wr, rx, nullptr, MMESS, Hdim, HP,
                                            HP, HP, Hdim, KT, nullptr, 0, 1, nullptr, 0);

  // ---- GRU message passing, DEPTH steps ----
  for (int d = 0; d < DEPTH_C; ++d) {
    gather_step<<<MMESS, blk128, 0, stream>>>(h, mess_graph, sumh_f, sumh_bf, hnei_bf);
    // z = sigmoid(zx + sum_h @ Wz_h^T)
    gemm_bf16_wmma<<<gM, blk256, 0, stream>>>(sumh_bf, Wz_h, z_buf, nullptr, MMESS, Hdim, HP,
                                              HP, HP, Hdim, KT, zx, Hdim, 1, nullptr, 1);
    // r = sigmoid(rx[row/K] + h_nei @ Ur^T + Ur_b)   (bf16 out, padded)
    gemm_bf16_wmma<<<gMK, blk256, 0, stream>>>(hnei_bf, Ur, nullptr, r_bf,
                                               MMESS * KNEI, Hdim, HP,
                                               HP, HP, HP, KT, rx, Hdim, KNEI, U_r_b, 1);
    sum_gh_kernel<<<MMESS, blk128, 0, stream>>>(r_bf, h, mess_graph, sumgh_bf);
    // pre_h = tanh(hx + sum_gh @ Wh_h^T)
    gemm_bf16_wmma<<<gM, blk256, 0, stream>>>(sumgh_bf, Wh_h, preh, nullptr, MMESS, Hdim, HP,
                                              HP, HP, Hdim, KT, hx, Hdim, 1, nullptr, 2);
    combine<<<(MMESS * Hdim + 255) / 256, 256, 0, stream>>>(h, z_buf, sumh_f, preh);
  }

  // ---- readout + discriminator ----
  node_agg<<<NNODES, blk128, 0, stream>>>(h, node_graph, mess_nei);
  root_pack<<<BROOT, blk128, 0, stream>>>(root_idx, fnode_e, mess_nei, root_bf);
  // h1 = leaky(root @ D1^T + b1)  -> bf16
  gemm_bf16_wmma<<<gB, blk256, 0, stream>>>(root_bf, D1_bf, nullptr, h1_bf, BROOT, DHdim, HP,
                                            K2, K2, HP, KT2, nullptr, 0, 1, D1_b, 3);
  // h2 = leaky(h1 @ D2^T + b2)  -> f32
  gemm_bf16_wmma<<<gB, blk256, 0, stream>>>(h1_bf, D2_bf, h2_f, nullptr, BROOT, DHdim, HP,
                                            HP, HP, DHdim, KT, nullptr, 0, 1, D2_b, 3);
  final_dot<<<BROOT, 32, 0, stream>>>(h2_f, D3_w, D3_b, (float*)d_out);
}
